// RelativeSelfAttentionLayer_56367150793014
// MI455X (gfx1250) — matrix operand order repaired and checked by reading, not compiled, not run on hardware
//
#include <hip/hip_runtime.h>
#include <hip/hip_bf16.h>
#include <cstdint>

// ---------------------------------------------------------------------------
// RelativeSelfAttentionLayer for MI455X (gfx1250, wave32, WMMA)
// B=4, C=512, T=1024, H=8, Dh=64.  All matmuls via v_wmma_f32_16x16x32_f16.
// ---------------------------------------------------------------------------

typedef __attribute__((ext_vector_type(16))) _Float16 v16h;
typedef __attribute__((ext_vector_type(8)))  float    v8f;

constexpr int BB   = 4;
constexpr int CC   = 512;
constexpr int TT   = 1024;
constexpr int HH   = 8;
constexpr int DH   = 64;
constexpr int SP   = 2 * TT - 1;   // 2047 relative positions
constexpr int SPAD = 2080;         // padded pos rows (mult of 32, >= 2063)
constexpr float NEGF  = -10000.0f;
constexpr float SCALE = 0.04419417382415922f; // 1/sqrt(512)

#if __has_builtin(__builtin_amdgcn_global_load_async_to_lds_b128)
#define HAVE_ASYNC_LDS 1
// Exact parameter types per hipcc diagnostic: int __vector_size__(16) pointers
// in global (addrspace 1) / LDS (addrspace 3) address spaces.
typedef int v4i __attribute__((__vector_size__(16)));
typedef __attribute__((address_space(1))) v4i* gas_v4i_p;
typedef __attribute__((address_space(3))) v4i* las_v4i_p;
#else
#define HAVE_ASYNC_LDS 0
#endif

__device__ __forceinline__ v8f wmma_f16(v16h a, v16h b, v8f c) {
  // D = A(16x32) * B(32x16) + C, f32 accumulate
  return __builtin_amdgcn_wmma_f32_16x16x32_f16(false, a, false, b, (short)0, c,
                                                false, false);
}

// A fragment (16x32 f16). lane: row = lane%16, half = lane/16.
// elements 0..7  <-> K = half*16 + e     (VGPR 0..3)
// elements 8..15 <-> K = half*16 + 8 + e (VGPR 4..7)
__device__ __forceinline__ v16h load_a_frag(const _Float16* base, int ld,
                                            int row, int half) {
  const _Float16* p = base + (size_t)row * ld + half * 16;
  v16h a;
#pragma unroll
  for (int e = 0; e < 8; ++e) { a[e] = p[e]; a[e + 8] = p[8 + e]; }
  return a;
}

// B fragment (32x16 f16), B[k][n] taken from a row-major source S[n][k]
// (k contiguous). lane: n = lane%16, half = lane/16; element e <-> K = half*16+e.
__device__ __forceinline__ v16h load_b_frag(const _Float16* base, int ld,
                                            int n, int half) {
  const _Float16* p = base + (size_t)n * ld + half * 16;
  v16h b;
#pragma unroll
  for (int e = 0; e < 16; ++e) b[e] = p[e];
  return b;
}

// ---------------------------------------------------------------------------
// Small utility kernels
// ---------------------------------------------------------------------------
__global__ void k_f32_to_f16(const float* __restrict__ s,
                             _Float16* __restrict__ d, int n) {
  int i = blockIdx.x * blockDim.x + threadIdx.x;
  if (i < n) d[i] = (_Float16)s[i];
}

// posT[j, c] = pos_emb[c, j]  (j < 2047, zero-padded to SPAD rows)
__global__ void k_cvt_pos(const float* __restrict__ pe,
                          _Float16* __restrict__ posT) {
  int i = blockIdx.x * blockDim.x + threadIdx.x;
  if (i >= SPAD * CC) return;
  int j = i / CC, c = i - j * CC;
  posT[i] = (j < SP) ? (_Float16)pe[(size_t)c * SP + j] : (_Float16)0.0f;
}

// Channel LayerNorm over C for each (b,t); writes token-major f16 [B*T, C].
// 32 tokens per block; lanes walk the T axis so global accesses coalesce.
__global__ void k_layernorm(const float* __restrict__ x,
                            const float* __restrict__ gamma,
                            const float* __restrict__ beta,
                            _Float16* __restrict__ xn) {
  int tok0 = blockIdx.x * 32;             // first token of this block
  int b = tok0 >> 10, t0 = tok0 & (TT - 1);
  int tid = threadIdx.x;                  // 256 threads
  int tt = tid & 31;                      // token within block
  int cc = tid >> 5;                      // channel group 0..7 (64 ch each)
  const float* xb = x + (size_t)b * CC * TT + t0 + tt;

  float s = 0.0f, s2 = 0.0f;
  for (int c = cc * 64; c < cc * 64 + 64; ++c) {
    float v = xb[(size_t)c * TT];
    s += v; s2 += v * v;
  }
  __shared__ float rs[8][32], rs2[8][32];
  rs[cc][tt] = s; rs2[cc][tt] = s2;
  __syncthreads();
  __shared__ float meanb[32], rstdb[32];
  if (cc == 0) {
    float S = 0.0f, S2 = 0.0f;
#pragma unroll
    for (int g = 0; g < 8; ++g) { S += rs[g][tt]; S2 += rs2[g][tt]; }
    float mean = S / (float)CC;
    meanb[tt] = mean;
    rstdb[tt] = rsqrtf(S2 / (float)CC - mean * mean + 1e-5f);
  }
  __syncthreads();
  __shared__ _Float16 tile[32][CC];       // 32 KB staging tile
  float mean = meanb[tt], rstd = rstdb[tt];
  for (int c = cc * 64; c < cc * 64 + 64; ++c) {
    float v = xb[(size_t)c * TT];         // L2-hot second read
    tile[tt][c] = (_Float16)((v - mean) * rstd * gamma[c] + beta[c]);
  }
  __syncthreads();
  // coalesced write-out: 16B chunks, contiguous per row
  for (int idx = tid; idx < 32 * (CC / 8); idx += 256) {
    int r = idx >> 6;                     // row (token)
    int ch = idx & 63;                    // 8-half chunk
    *(uint4*)&xn[((size_t)tok0 + r) * CC + ch * 8] =
        *(const uint4*)&tile[r][ch * 8];
  }
}

// qu = q + u_bias[o], qv = q + v_bias[o]; vT[b,h,d,t] = v[token, h*64+d]
__global__ void k_post_qkv(const _Float16* __restrict__ q,
                           const _Float16* __restrict__ v,
                           const float* __restrict__ ubias,
                           const float* __restrict__ vbias,
                           _Float16* __restrict__ qu, _Float16* __restrict__ qv,
                           _Float16* __restrict__ vT) {
  int i = blockIdx.x * blockDim.x + threadIdx.x;
  if (i >= BB * TT * CC) return;
  int o = i & (CC - 1);
  int m = i >> 9;                  // token id
  int b = m >> 10, t = m & (TT - 1);
  float qf = (float)q[i];
  qu[i] = (_Float16)(qf + ubias[o]);
  qv[i] = (_Float16)(qf + vbias[o]);
  int h = o >> 6, d = o & 63;
  vT[(((size_t)(b * HH + h)) * DH + d) * TT + t] = v[i];
}

// ---------------------------------------------------------------------------
// WMMA GEMM, 32x32 tile per wave (2x2 WMMA tiles, register-level A/B reuse):
// Out[m,n](f16) = sum_k A[m,k]*W[n,k] + bias[n]
// ---------------------------------------------------------------------------
__global__ void k_gemm32(const _Float16* __restrict__ A,
                         const _Float16* __restrict__ W,
                         const float* __restrict__ bias,
                         _Float16* __restrict__ Out, int M, int N, int K) {
  int wave = (blockIdx.x * blockDim.x + threadIdx.x) >> 5;
  int lane = threadIdx.x & 31;
  int tilesN = N >> 5;
  int nt = wave % tilesN;
  int mt = wave / tilesN;
  if (mt >= (M >> 5)) return;      // wave-uniform; EXEC stays all-ones
  int row = lane & 15, half = lane >> 4;
  const _Float16* A0 = A + (size_t)(mt * 32) * K;
  const _Float16* A1 = A0 + (size_t)16 * K;
  const _Float16* W0 = W + (size_t)(nt * 32) * K;
  const _Float16* W1 = W0 + (size_t)16 * K;
  v8f acc00 = {}, acc01 = {}, acc10 = {}, acc11 = {};
  for (int k = 0; k < K; k += 32) {
    v16h a0 = load_a_frag(A0 + k, K, row, half);
    v16h a1 = load_a_frag(A1 + k, K, row, half);
    v16h b0 = load_b_frag(W0 + k, K, row, half);
    v16h b1 = load_b_frag(W1 + k, K, row, half);
    acc00 = wmma_f16(a0, b0, acc00);
    acc01 = wmma_f16(a0, b1, acc01);
    acc10 = wmma_f16(a1, b0, acc10);
    acc11 = wmma_f16(a1, b1, acc11);
  }
  int n0 = nt * 32 + row;
  float bv0 = bias ? bias[n0]      : 0.0f;
  float bv1 = bias ? bias[n0 + 16] : 0.0f;
#pragma unroll
  for (int r = 0; r < 8; ++r) {
    int m0 = mt * 32 + r + 8 * half;
    Out[(size_t)m0 * N + n0]             = (_Float16)(acc00[r] + bv0);
    Out[(size_t)m0 * N + n0 + 16]        = (_Float16)(acc01[r] + bv1);
    Out[(size_t)(m0 + 16) * N + n0]      = (_Float16)(acc10[r] + bv0);
    Out[(size_t)(m0 + 16) * N + n0 + 16] = (_Float16)(acc11[r] + bv1);
  }
}

// Final projection (32x32/wave): f32 output scattered to [B, C, T] layout.
__global__ void k_gemm_out32(const _Float16* __restrict__ A,
                             const _Float16* __restrict__ W,
                             const float* __restrict__ bias,
                             float* __restrict__ out, int M, int N, int K) {
  int wave = (blockIdx.x * blockDim.x + threadIdx.x) >> 5;
  int lane = threadIdx.x & 31;
  int tilesN = N >> 5;
  int nt = wave % tilesN;
  int mt = wave / tilesN;
  if (mt >= (M >> 5)) return;
  int row = lane & 15, half = lane >> 4;
  const _Float16* A0 = A + (size_t)(mt * 32) * K;
  const _Float16* A1 = A0 + (size_t)16 * K;
  const _Float16* W0 = W + (size_t)(nt * 32) * K;
  const _Float16* W1 = W0 + (size_t)16 * K;
  v8f acc00 = {}, acc01 = {}, acc10 = {}, acc11 = {};
  for (int k = 0; k < K; k += 32) {
    v16h a0 = load_a_frag(A0 + k, K, row, half);
    v16h a1 = load_a_frag(A1 + k, K, row, half);
    v16h b0 = load_b_frag(W0 + k, K, row, half);
    v16h b1 = load_b_frag(W1 + k, K, row, half);
    acc00 = wmma_f16(a0, b0, acc00);
    acc01 = wmma_f16(a0, b1, acc01);
    acc10 = wmma_f16(a1, b0, acc10);
    acc11 = wmma_f16(a1, b1, acc11);
  }
  int n0 = nt * 32 + row;
  float bv0 = bias[n0], bv1 = bias[n0 + 16];
#pragma unroll
  for (int r = 0; r < 8; ++r) {
    int m0 = mt * 32 + r + 8 * half;          // token = b*T + t
    int b0i = m0 >> 10, t0i = m0 & (TT - 1);
    int b1i = (m0 + 16) >> 10, t1i = (m0 + 16) & (TT - 1);
    out[((size_t)b0i * N + n0) * TT + t0i]        = acc00[r] + bv0;
    out[((size_t)b0i * N + n0 + 16) * TT + t0i]   = acc01[r] + bv1;
    out[((size_t)b1i * N + n0) * TT + t1i]        = acc10[r] + bv0;
    out[((size_t)b1i * N + n0 + 16) * TT + t1i]   = acc11[r] + bv1;
  }
}

// ---------------------------------------------------------------------------
// Attention pass 1: per-query-row online softmax stats (m_i, Z_i).
// score[i,j] = (qu_i . k_j + pos_raw[i, T-1-i+j]) / sqrt(C)   (rel_shift fused)
// One wave per (b,h, 16-row i-tile).
// ---------------------------------------------------------------------------
__global__ void k_attn_stats(const _Float16* __restrict__ qu,
                             const _Float16* __restrict__ qv,
                             const _Float16* __restrict__ kh,
                             const _Float16* __restrict__ pT,
                             const float* __restrict__ xmask,
                             float* __restrict__ mbuf, float* __restrict__ zbuf) {
  const int ntile = TT >> 4;
  int bh = blockIdx.x / ntile;
  int it = blockIdx.x - bh * ntile;
  int b = bh >> 3, h = bh & 7;
  int i0 = it << 4;
  int lane = threadIdx.x & 31, row = lane & 15, half = lane >> 4;
  const size_t tok = (size_t)b * TT;
  const int co = h * DH;

  v16h aqu0 = load_a_frag(qu + (tok + i0) * CC + co,      CC, row, half);
  v16h aqu1 = load_a_frag(qu + (tok + i0) * CC + co + 32, CC, row, half);
  v16h aqv0 = load_a_frag(qv + (tok + i0) * CC + co,      CC, row, half);
  v16h aqv1 = load_a_frag(qv + (tok + i0) * CC + co + 32, CC, row, half);

  float mrow[8];
#pragma unroll
  for (int r = 0; r < 8; ++r) mrow[r] = xmask[b * TT + i0 + r + 8 * half];

  __shared__ float poslds[16][48];

  float mrun[8], zrun[8];
#pragma unroll
  for (int r = 0; r < 8; ++r) { mrun[r] = -1e30f; zrun[r] = 0.0f; }

  for (int j0 = 0; j0 < TT; j0 += 16) {
    v16h bk0 = load_b_frag(kh + (tok + j0) * CC + co,      CC, row, half);
    v16h bk1 = load_b_frag(kh + (tok + j0) * CC + co + 32, CC, row, half);
    v8f zero = {};
    v8f sc = wmma_f16(aqu0, bk0, zero);
    sc = wmma_f16(aqu1, bk1, sc);

    int base = TT - 16 - i0 + j0;      // min needed pos column
    int jjb  = base & ~15;             // aligned band start, band width 48
#pragma unroll
    for (int tt3 = 0; tt3 < 3; ++tt3) {
      v16h bp0 = load_b_frag(pT + (size_t)(jjb + tt3 * 16) * CC + co,      CC, row, half);
      v16h bp1 = load_b_frag(pT + (size_t)(jjb + tt3 * 16) * CC + co + 32, CC, row, half);
      v8f pd = wmma_f16(aqv0, bp0, zero);
      pd = wmma_f16(aqv1, bp1, pd);
#pragma unroll
      for (int r = 0; r < 8; ++r) poslds[r + 8 * half][tt3 * 16 + row] = pd[r];
    }
    __syncthreads();

    float mcol = xmask[b * TT + j0 + row];
#pragma unroll
    for (int r = 0; r < 8; ++r) {
      int i = i0 + r + 8 * half;
      int off = (TT - 1 - i + j0 + row) - jjb;       // rel_shift gather, in [0,48)
      float s = (sc[r] + poslds[r + 8 * half][off]) * SCALE;
      if (mrow[r] * mcol == 0.0f) s = NEGF;
      float mx = s;                                   // row max over 16 lanes
#pragma unroll
      for (int d = 1; d < 16; d <<= 1) mx = fmaxf(mx, __shfl_xor(mx, d, 32));
      float mnew = fmaxf(mrun[r], mx);
      float e = __expf(s - mnew);
#pragma unroll
      for (int d = 1; d < 16; d <<= 1) e += __shfl_xor(e, d, 32);
      zrun[r] = zrun[r] * __expf(mrun[r] - mnew) + e;
      mrun[r] = mnew;
    }
    __syncthreads();
  }
  if (row == 0) {
#pragma unroll
    for (int r = 0; r < 8; ++r) {
      int i = i0 + r + 8 * half;
      mbuf[(size_t)bh * TT + i] = mrun[r];
      zbuf[(size_t)bh * TT + i] = zrun[r];
    }
  }
}

// ---------------------------------------------------------------------------
// Attention pass 2: ctx[j,d] = sum_i (exp(score[i,j]-m_i)/Z_i) * V[i,d]
// (reference applies attention transposed). One wave per (b,h, 16-col j-tile).
// V tile staged into LDS via CDNA5 async-to-LDS DMA (overlaps score WMMAs);
// P-hat tiles transposed through LDS into WMMA A fragments.
// ---------------------------------------------------------------------------
__global__ void k_attn_ctx(const _Float16* __restrict__ qu,
                           const _Float16* __restrict__ qv,
                           const _Float16* __restrict__ kh,
                           const _Float16* __restrict__ pT,
                           const _Float16* __restrict__ vT,
                           const float* __restrict__ xmask,
                           const float* __restrict__ mbuf,
                           const float* __restrict__ zbuf,
                           _Float16* __restrict__ ctx) {
  const int ntile = TT >> 4;
  int bh = blockIdx.x / ntile;
  int jt = blockIdx.x - bh * ntile;
  int b = bh >> 3, h = bh & 7;
  int j0 = jt << 4;
  int lane = threadIdx.x & 31, row = lane & 15, half = lane >> 4;
  const size_t tok = (size_t)b * TT;
  const int co = h * DH;

  v16h bk0 = load_b_frag(kh + (tok + j0) * CC + co,      CC, row, half);
  v16h bk1 = load_b_frag(kh + (tok + j0) * CC + co + 32, CC, row, half);
  float mcol = xmask[b * TT + j0 + row];

  __shared__ float    poslds[16][48];
  __shared__ _Float16 plds[32][20];   // padded stride to soften bank conflicts
  __shared__ _Float16 vlds[64][32];   // V tile: 64 d-rows x 32 i-cols

  v8f acc[4] = {{}, {}, {}, {}};
  const _Float16* vbase = vT + (size_t)bh * DH * TT;

  for (int i0 = 0; i0 < TT; i0 += 32) {
    // ---- kick async stage of V tile (64 x 32 f16 = 4KB) into LDS ----------
#pragma unroll
    for (int inst = 0; inst < 8; ++inst) {
      int c = inst * 32 + lane;           // 16B chunk id, 4 chunks per d-row
      int rowd = c >> 2, sub = c & 3;
      const _Float16* g = vbase + (size_t)rowd * TT + i0 + sub * 8;
      _Float16* l = &vlds[rowd][sub * 8];
#if HAVE_ASYNC_LDS
      __builtin_amdgcn_global_load_async_to_lds_b128(
          (gas_v4i_p)(uintptr_t)g, (las_v4i_p)(uint32_t)(uintptr_t)l, 0, 0);
#else
      *(uint4*)l = *(const uint4*)g;
#endif
    }
    // ---- score tiles + P-hat while the DMA is in flight --------------------
#pragma unroll
    for (int sub = 0; sub < 2; ++sub) {
      int is0 = i0 + sub * 16;
      v16h aqu0 = load_a_frag(qu + (tok + is0) * CC + co,      CC, row, half);
      v16h aqu1 = load_a_frag(qu + (tok + is0) * CC + co + 32, CC, row, half);
      v16h aqv0 = load_a_frag(qv + (tok + is0) * CC + co,      CC, row, half);
      v16h aqv1 = load_a_frag(qv + (tok + is0) * CC + co + 32, CC, row, half);
      v8f zero = {};
      v8f sc = wmma_f16(aqu0, bk0, zero);
      sc = wmma_f16(aqu1, bk1, sc);
      int base = TT - 16 - is0 + j0;
      int jjb  = base & ~15;
#pragma unroll
      for (int tt3 = 0; tt3 < 3; ++tt3) {
        v16h bp0 = load_b_frag(pT + (size_t)(jjb + tt3 * 16) * CC + co,      CC, row, half);
        v16h bp1 = load_b_frag(pT + (size_t)(jjb + tt3 * 16) * CC + co + 32, CC, row, half);
        v8f pd = wmma_f16(aqv0, bp0, zero);
        pd = wmma_f16(aqv1, bp1, pd);
#pragma unroll
        for (int r = 0; r < 8; ++r) poslds[r + 8 * half][tt3 * 16 + row] = pd[r];
      }
      __syncthreads();
#pragma unroll
      for (int r = 0; r < 8; ++r) {
        int i = is0 + r + 8 * half;
        int off = (TT - 1 - i + j0 + row) - jjb;
        float s = (sc[r] + poslds[r + 8 * half][off]) * SCALE;
        float mr = xmask[b * TT + i];
        if (mr * mcol == 0.0f) s = NEGF;
        float mi = mbuf[(size_t)bh * TT + i];
        float zi = zbuf[(size_t)bh * TT + i];
        float pv = __expf(s - mi) / zi;
        plds[sub * 16 + r + 8 * half][row] = (_Float16)pv;   // P-hat[i][j]
      }
      __syncthreads();
    }
    // ---- drain async DMA, then context WMMAs from LDS ----------------------
#if HAVE_ASYNC_LDS
#if __has_builtin(__builtin_amdgcn_s_wait_asynccnt)
    __builtin_amdgcn_s_wait_asynccnt(0);
#else
    asm volatile("s_wait_asynccnt 0x0" ::: "memory");
#endif
#endif
    __syncthreads();
    // A fragment (rows = j, K = i offset 0..31) read transposed from LDS
    v16h af;
#pragma unroll
    for (int e = 0; e < 8; ++e) {
      af[e]     = plds[half * 16 + e][row];
      af[e + 8] = plds[half * 16 + 8 + e][row];
    }
#pragma unroll
    for (int dt = 0; dt < 4; ++dt) {
      v16h bv;
#pragma unroll
      for (int e = 0; e < 16; ++e) bv[e] = vlds[dt * 16 + row][half * 16 + e];
      acc[dt] = wmma_f16(af, bv, acc[dt]);
    }
    __syncthreads();
  }
#pragma unroll
  for (int dt = 0; dt < 4; ++dt) {
#pragma unroll
    for (int r = 0; r < 8; ++r) {
      int j = j0 + r + 8 * half;
      ctx[(tok + j) * CC + co + dt * 16 + row] = (_Float16)acc[dt][r];
    }
  }
}

// ---------------------------------------------------------------------------
// Host-side launch
// ---------------------------------------------------------------------------
extern "C" void kernel_launch(void* const* d_in, const int* in_sizes, int n_in,
                              void* d_out, int out_size, void* d_ws, size_t ws_size,
                              hipStream_t stream) {
  (void)in_sizes; (void)n_in; (void)out_size; (void)ws_size;
  const float* x      = (const float*)d_in[0];
  const float* posemb = (const float*)d_in[1];
  const float* xmask  = (const float*)d_in[2];
  const float* gamma  = (const float*)d_in[3];
  const float* beta   = (const float*)d_in[4];
  const float* Wq     = (const float*)d_in[5];
  const float* biasq  = (const float*)d_in[6];
  const float* Wk     = (const float*)d_in[7];
  const float* biask  = (const float*)d_in[8];
  const float* Wv     = (const float*)d_in[9];
  const float* biasv  = (const float*)d_in[10];
  const float* Wp     = (const float*)d_in[11];
  const float* ubias  = (const float*)d_in[12];
  const float* vbias  = (const float*)d_in[13];
  const float* Wo     = (const float*)d_in[14];
  const float* biaso  = (const float*)d_in[15];
  float* out = (float*)d_out;

  char* ws = (char*)d_ws;
  size_t off = 0;
  auto alloc = [&](size_t bytes) {
    char* p = ws + off;
    off = (off + bytes + 255) & ~(size_t)255;
    return p;
  };
  const size_t WSZ  = (size_t)CC * CC * sizeof(_Float16);
  const size_t ACT  = (size_t)BB * TT * CC * sizeof(_Float16);
  const size_t PSZ  = (size_t)SPAD * CC * sizeof(_Float16);
  _Float16* wq_h   = (_Float16*)alloc(WSZ);
  _Float16* wk_h   = (_Float16*)alloc(WSZ);
  _Float16* wv_h   = (_Float16*)alloc(WSZ);
  _Float16* wp_h   = (_Float16*)alloc(WSZ);
  _Float16* wo_h   = (_Float16*)alloc(WSZ);
  _Float16* xn_h   = (_Float16*)alloc(ACT);
  _Float16* q_h    = (_Float16*)alloc(ACT);
  _Float16* k_h    = (_Float16*)alloc(ACT);
  _Float16* v_h    = (_Float16*)alloc(ACT);
  _Float16* qu_h   = (_Float16*)alloc(ACT);
  _Float16* qv_h   = (_Float16*)alloc(ACT);
  _Float16* vT_h   = (_Float16*)alloc(ACT);
  _Float16* posT_h = (_Float16*)alloc(PSZ);
  _Float16* pT_h   = (_Float16*)alloc(PSZ);
  _Float16* ctx_h  = (_Float16*)alloc(ACT);
  float*    mbuf   = (float*)alloc((size_t)BB * HH * TT * sizeof(float));
  float*    zbuf   = (float*)alloc((size_t)BB * HH * TT * sizeof(float));

  // 1) weight / pos conversion to f16
  const int nw = CC * CC;
  k_f32_to_f16<<<(nw + 255) / 256, 256, 0, stream>>>(Wq, wq_h, nw);
  k_f32_to_f16<<<(nw + 255) / 256, 256, 0, stream>>>(Wk, wk_h, nw);
  k_f32_to_f16<<<(nw + 255) / 256, 256, 0, stream>>>(Wv, wv_h, nw);
  k_f32_to_f16<<<(nw + 255) / 256, 256, 0, stream>>>(Wp, wp_h, nw);
  k_f32_to_f16<<<(nw + 255) / 256, 256, 0, stream>>>(Wo, wo_h, nw);
  const int npos = SPAD * CC;
  k_cvt_pos<<<(npos + 255) / 256, 256, 0, stream>>>(posemb, posT_h);

  // 2) channel LayerNorm -> token-major f16 (coalesced, LDS-staged)
  k_layernorm<<<BB * TT / 32, 256, 0, stream>>>(x, gamma, beta, xn_h);

  // 3) projections (WMMA GEMMs, 32x32 tile per wave)
  const int w_qkv = (BB * TT / 32) * (CC / 32);   // 2048 waves -> 256 blocks
  const int w_p   = (SPAD / 32) * (CC / 32);      // 1040 waves -> 130 blocks
  k_gemm32<<<w_qkv / 8, 256, 0, stream>>>(xn_h, wq_h, biasq, q_h, BB * TT, CC, CC);
  k_gemm32<<<w_qkv / 8, 256, 0, stream>>>(xn_h, wk_h, biask, k_h, BB * TT, CC, CC);
  k_gemm32<<<w_qkv / 8, 256, 0, stream>>>(xn_h, wv_h, biasv, v_h, BB * TT, CC, CC);
  k_gemm32<<<w_p / 8, 256, 0, stream>>>(posT_h, wp_h, nullptr, pT_h, SPAD, CC, CC);

  // 4) add u/v head biases, build transposed V
  const int nelem = BB * TT * CC;
  k_post_qkv<<<(nelem + 255) / 256, 256, 0, stream>>>(q_h, v_h, ubias, vbias,
                                                      qu_h, qv_h, vT_h);

  // 5) fused attention (two-pass flash, rel_shift fused, softmax stats in f32)
  const int ablocks = BB * HH * (TT / 16);   // 2048 single-wave workgroups
  k_attn_stats<<<ablocks, 32, 0, stream>>>(qu_h, qv_h, k_h, pT_h, xmask, mbuf, zbuf);
  k_attn_ctx<<<ablocks, 32, 0, stream>>>(qu_h, qv_h, k_h, pT_h, vT_h, xmask,
                                         mbuf, zbuf, ctx_h);

  // 6) output projection, scatter to [B, C, T] f32
  k_gemm_out32<<<w_qkv / 8, 256, 0, stream>>>(ctx_h, wo_h, biaso, out, BB * TT, CC, CC);
}
